// Predictor_55533927137951
// MI455X (gfx1250) — matrix operand order repaired and checked by reading, not compiled, not run on hardware
//
#include <hip/hip_runtime.h>
#include <hip/hip_bf16.h>

typedef __attribute__((ext_vector_type(16))) _Float16 v16h;
typedef __attribute__((ext_vector_type(8)))  float    v8f;

#define BB   16
#define LL   1024
#define CC   32
#define HH   2
#define DD   32      // per-head dim
#define HD   (HH*DD) // 64
#define NBLK 2
#define POSD 32

__device__ __forceinline__ int ak(int e, int g) {
    // contiguous k order: k = 16*g + e (matches B operand fills)
    return e + g * 16;
}

__device__ __forceinline__ v8f wmma_f16(v16h a, v16h b, v8f c) {
    return __builtin_amdgcn_wmma_f32_16x16x32_f16(false, a, false, b, (short)0, c, false, false);
}

// ---------------- pos = cumsum(mask) - 1, per batch ----------------
__global__ void pos_kernel(const unsigned char* __restrict__ mask, int* __restrict__ pos) {
    int b = threadIdx.x;
    int c = 0;
    for (int l = 0; l < LL; ++l) {
        c += mask[b * LL + l] ? 1 : 0;
        pos[b * LL + l] = c - 1;
    }
}

// ---------------- f32 -> f16 cast ----------------
__global__ void cast_kernel(const float* __restrict__ x, _Float16* __restrict__ y, int n) {
    int i = blockIdx.x * blockDim.x + threadIdx.x;
    if (i < n) y[i] = (_Float16)x[i];
}

// ---------------- QKV projection: [16384,32] x [32,64] per weight ----------------
// one wave = one 16x16 output tile; K=32 -> single WMMA
__global__ __launch_bounds__(128) void qkv_kernel(
    const _Float16* __restrict__ x,
    const float* __restrict__ Wq, const float* __restrict__ Wk, const float* __restrict__ Wv,
    _Float16* __restrict__ q, _Float16* __restrict__ k, _Float16* __restrict__ v)
{
    const int tid = threadIdx.x, wid = tid >> 5, lane = tid & 31;
    const int n = lane & 15, g = lane >> 4, m = n;
    const int t = blockIdx.x * 4 + wid;              // 0..12287
    const int mt = t / 12, rem = t % 12;
    const int wsel = rem >> 2, nt = rem & 3;

    const float*    W   = (wsel == 0) ? Wq : (wsel == 1) ? Wk : Wv;
    _Float16*       out = (wsel == 0) ? q  : (wsel == 1) ? k  : v;

    v16h a, bf;
#pragma unroll
    for (int e = 0; e < 16; ++e)
        a[e] = x[(size_t)(mt * 16 + m) * CC + ak(e, g)];
#pragma unroll
    for (int e = 0; e < 16; ++e)
        bf[e] = (_Float16)W[(size_t)(g * 16 + e) * HD + nt * 16 + n];

    v8f c = {};
    c = wmma_f16(a, bf, c);
#pragma unroll
    for (int r = 0; r < 8; ++r)
        out[(size_t)(mt * 16 + r + 8 * g) * HD + nt * 16 + n] = (_Float16)c[r];
}

// ---------------- fused attention: scores + bias + softmax + A@V ----------------
// one workgroup (8 waves) per (b, h, 16-row tile); each wave owns 8 column tiles.
__global__ __launch_bounds__(256) void attn_kernel(
    const _Float16* __restrict__ q, const _Float16* __restrict__ k,
    const _Float16* __restrict__ v, const int* __restrict__ pos,
    const unsigned char* __restrict__ mask, const float* __restrict__ Wb,
    float* __restrict__ attn, _Float16* __restrict__ o_out, int blk)
{
    __shared__ _Float16 a_lds[16 * LL];   // probabilities, f16, 32 KB
    __shared__ float    red[8 * 16];      // per-wave row partials
    __shared__ float    o_lds[16 * DD];   // O accumulator
    __shared__ float    wbh[POSD];        // W_b column for this head

    const float SCALE = 0.17677669529663687f;  // 1/sqrt(32)
    const float NEGINF = -1e22f;

    const int tid = threadIdx.x, wid = tid >> 5, lane = tid & 31;
    const int n = lane & 15, g = lane >> 4, m = n;

    const int it = blockIdx.x & 63;
    const int h  = (blockIdx.x >> 6) & (HH - 1);
    const int b  = blockIdx.x >> 7;

    o_lds[tid] = 0.f; o_lds[tid + 256] = 0.f;
    if (tid < POSD) wbh[tid] = Wb[tid * HH + h];
    __syncthreads();

    // ---- A fragment of Q (reused for all 8 column tiles) ----
    const size_t qrow = ((size_t)b * LL + it * 16 + m) * HD + h * DD;
    v16h aq;
#pragma unroll
    for (int e = 0; e < 16; ++e) aq[e] = q[qrow + ak(e, g)];

    int pi[8]; int mi[8];
#pragma unroll
    for (int r = 0; r < 8; ++r) {
        int row = it * 16 + r + 8 * g;
        pi[r] = pos[b * LL + row];
        mi[r] = mask[b * LL + row];
    }

    // ---- scores for this wave's 8 column tiles ----
    float s[8][8];
#pragma unroll
    for (int t = 0; t < 8; ++t) {
        const int jb = (wid * 8 + t) * 16;
        const size_t krow = ((size_t)b * LL + jb + n) * HD + h * DD;
        v16h bk;
#pragma unroll
        for (int e = 0; e < 16; ++e) bk[e] = k[krow + g * 16 + e];
        v8f c = {};
        c = wmma_f16(aq, bk, c);
        const int pj = pos[b * LL + jb + n];
        const int mj = mask[b * LL + jb + n];
#pragma unroll
        for (int r = 0; r < 8; ++r) {
            int idx = pi[r] - pj + POSD / 2;
            idx = idx < 0 ? 0 : (idx > POSD - 1 ? POSD - 1 : idx);
            float bias = (mi[r] && mj) ? wbh[idx] : NEGINF;
            s[t][r] = c[r] * SCALE + bias;
        }
    }

    // ---- row max (shuffle within 16-lane half, then cross-wave via LDS) ----
    float rm[8];
#pragma unroll
    for (int r = 0; r < 8; ++r) {
        float mv = s[0][r];
#pragma unroll
        for (int t = 1; t < 8; ++t) mv = fmaxf(mv, s[t][r]);
#pragma unroll
        for (int off = 8; off >= 1; off >>= 1) mv = fmaxf(mv, __shfl_xor(mv, off, 32));
        rm[r] = mv;
    }
    if (n == 0) {
#pragma unroll
        for (int r = 0; r < 8; ++r) red[wid * 16 + r + 8 * g] = rm[r];
    }
    __syncthreads();
#pragma unroll
    for (int r = 0; r < 8; ++r) {
        const int row = r + 8 * g;
        float mv = red[row];
#pragma unroll
        for (int w = 1; w < 8; ++w) mv = fmaxf(mv, red[w * 16 + row]);
        rm[r] = mv;
    }
    __syncthreads();

    // ---- exp + row sum ----
    float rs[8];
#pragma unroll
    for (int r = 0; r < 8; ++r) {
        float sum = 0.f;
#pragma unroll
        for (int t = 0; t < 8; ++t) {
            float e = __expf(s[t][r] - rm[r]);
            s[t][r] = e;
            sum += e;
        }
#pragma unroll
        for (int off = 8; off >= 1; off >>= 1) sum += __shfl_xor(sum, off, 32);
        rs[r] = sum;
    }
    if (n == 0) {
#pragma unroll
        for (int r = 0; r < 8; ++r) red[wid * 16 + r + 8 * g] = rs[r];
    }
    __syncthreads();
#pragma unroll
    for (int r = 0; r < 8; ++r) {
        const int row = r + 8 * g;
        float sum = 0.f;
#pragma unroll
        for (int w = 0; w < 8; ++w) sum += red[w * 16 + row];
        rs[r] = 1.f / sum;
    }

    // ---- normalize: write attention (f32, only write of the maps) + stash f16 in LDS ----
    const size_t abase = (((size_t)b * NBLK + blk) * HH + h) * (size_t)LL * LL;
#pragma unroll
    for (int t = 0; t < 8; ++t) {
        const int jb = (wid * 8 + t) * 16;
#pragma unroll
        for (int r = 0; r < 8; ++r) {
            const int row = r + 8 * g;
            float a = s[t][r] * rs[r];
            attn[abase + (size_t)(it * 16 + row) * LL + jb + n] = a;
            a_lds[row * LL + jb + n] = (_Float16)a;
        }
    }
    __syncthreads();

    // ---- O = A @ V : each wave covers K = [wid*128, wid*128+128), 4 WMMAs x 2 d-tiles ----
    v8f acc0 = {}, acc1 = {};
#pragma unroll
    for (int ks = 0; ks < 4; ++ks) {
        const int kb = wid * 128 + ks * 32;
        v16h af;
#pragma unroll
        for (int e = 0; e < 16; ++e) af[e] = a_lds[m * LL + kb + ak(e, g)];
        const size_t vrow = ((size_t)b * LL + kb + g * 16) * HD + h * DD;
        v16h b0, b1;
#pragma unroll
        for (int e = 0; e < 16; ++e) {
            b0[e] = v[vrow + (size_t)e * HD + n];
            b1[e] = v[vrow + (size_t)e * HD + 16 + n];
        }
        acc0 = wmma_f16(af, b0, acc0);
        acc1 = wmma_f16(af, b1, acc1);
    }
#pragma unroll
    for (int r = 0; r < 8; ++r) {
        const int row = r + 8 * g;
        atomicAdd(&o_lds[row * DD + n], acc0[r]);
        atomicAdd(&o_lds[row * DD + 16 + n], acc1[r]);
    }
    __syncthreads();

    // ---- write O (f16) ----
    {
        const int i0 = tid, i1 = tid + 256;
        o_out[((size_t)b * LL + it * 16 + i0 / DD) * HD + h * DD + (i0 % DD)] = (_Float16)o_lds[i0];
        o_out[((size_t)b * LL + it * 16 + i1 / DD) * HD + h * DD + (i1 % DD)] = (_Float16)o_lds[i1];
    }
}

// ---------------- output projection: [16384,64] x [64,32] ----------------
__global__ __launch_bounds__(128) void proj_kernel(
    const _Float16* __restrict__ o, const float* __restrict__ Wo,
    float* __restrict__ xout, _Float16* __restrict__ x16)
{
    const int tid = threadIdx.x, wid = tid >> 5, lane = tid & 31;
    const int n = lane & 15, g = lane >> 4, m = n;
    const int t = blockIdx.x * 4 + wid;          // 0..2047
    const int mt = t >> 1, nt = t & 1;

    v8f c = {};
#pragma unroll
    for (int ks = 0; ks < 2; ++ks) {
        v16h a, bf;
#pragma unroll
        for (int e = 0; e < 16; ++e)
            a[e] = o[(size_t)(mt * 16 + m) * HD + ks * 32 + ak(e, g)];
#pragma unroll
        for (int e = 0; e < 16; ++e)
            bf[e] = (_Float16)Wo[(size_t)(ks * 32 + g * 16 + e) * CC + nt * 16 + n];
        c = wmma_f16(a, bf, c);
    }
#pragma unroll
    for (int r = 0; r < 8; ++r) {
        const size_t idx = (size_t)(mt * 16 + r + 8 * g) * CC + nt * 16 + n;
        xout[idx] = c[r];
        x16[idx] = (_Float16)c[r];
    }
}

extern "C" void kernel_launch(void* const* d_in, const int* in_sizes, int n_in,
                              void* d_out, int out_size, void* d_ws, size_t ws_size,
                              hipStream_t stream) {
    (void)in_sizes; (void)n_in; (void)out_size; (void)ws_size;
    const float*         x_in = (const float*)d_in[0];
    const unsigned char* mask = (const unsigned char*)d_in[1];  // numpy bool, 1 byte
    const float* Wq = (const float*)d_in[2];
    const float* Wk = (const float*)d_in[3];
    const float* Wv = (const float*)d_in[4];
    const float* Wo = (const float*)d_in[5];
    const float* Wb = (const float*)d_in[6];

    float* out_x = (float*)d_out;
    float* attn  = out_x + (size_t)BB * LL * CC;   // [B, nblock, H, L, L]

    // workspace layout (~9.1 MB)
    char* ws = (char*)d_ws;
    int*       pos = (int*)ws;        ws += (size_t)BB * LL * sizeof(int);
    _Float16*  x16 = (_Float16*)ws;   ws += (size_t)BB * LL * CC * sizeof(_Float16);
    _Float16*  q   = (_Float16*)ws;   ws += (size_t)BB * LL * HD * sizeof(_Float16);
    _Float16*  k   = (_Float16*)ws;   ws += (size_t)BB * LL * HD * sizeof(_Float16);
    _Float16*  v   = (_Float16*)ws;   ws += (size_t)BB * LL * HD * sizeof(_Float16);
    _Float16*  o16 = (_Float16*)ws;   ws += (size_t)BB * LL * HD * sizeof(_Float16);

    pos_kernel<<<1, BB, 0, stream>>>(mask, pos);
    const int nx = BB * LL * CC;
    cast_kernel<<<(nx + 255) / 256, 256, 0, stream>>>(x_in, x16, nx);

    for (int blk = 0; blk < NBLK; ++blk) {
        // 1024 m-tiles * 3 weights * 4 n-tiles = 12288 waves, 4 waves/block
        qkv_kernel<<<3072, 128, 0, stream>>>(x16, Wq, Wk, Wv, q, k, v);
        // one workgroup per (b, h, 16-row tile)
        attn_kernel<<<BB * HH * (LL / 16), 256, 0, stream>>>(q, k, v, pos, mask, Wb,
                                                             attn, o16, blk);
        // 1024 m-tiles * 2 n-tiles = 2048 waves
        proj_kernel<<<512, 128, 0, stream>>>(o16, Wo, out_x, x16);
    }
}
